// TemporalMambaEncoder_1846835938145
// MI455X (gfx1250) — matrix operand order repaired and checked by reading, not compiled, not run on hardware
//
#include <hip/hip_runtime.h>
#include <hip/hip_bf16.h>
#include <math.h>

// ---------------- problem constants ----------------
constexpr int LNUM   = 4;
constexpr int DMODEL = 256;
constexpr int DINNER = 512;
constexpr int NST    = 16;
constexpr int KCONV  = 4;
constexpr int DTRANK = 16;
constexpr int BATCH  = 4;
constexpr int TLEN   = 1024;
constexpr int ROWS   = BATCH * TLEN;       // 4096 token rows
constexpr int MT     = ROWS / 16;          // 256 M-tiles

typedef __bf16 bf16;
typedef __attribute__((ext_vector_type(16))) __bf16 v16bf;
typedef __attribute__((ext_vector_type(8)))  __bf16 v8bf;
typedef __attribute__((ext_vector_type(8)))  float  v8f;

__device__ __forceinline__ float sigmoidf_(float x) { return 1.0f / (1.0f + __expf(-x)); }

// ---------------- weight pre-pack into WMMA B-fragment layout ----------------
// out[((kt*nTiles + nt)*32 + lane)*16 + i] = W[kt*32 + (lane<16?0:16) + i][nt*16 + lane%16]
__global__ void pack_b_kernel(const float* __restrict__ W, bf16* __restrict__ out,
                              int Kd, int Nd, int ktiles) {
    int idx = blockIdx.x * blockDim.x + threadIdx.x;
    int total = ktiles * 32 * Nd;
    if (idx >= total) return;
    int i    = idx & 15;
    int lane = (idx >> 4) & 31;
    int tile = idx >> 9;
    int nTiles = Nd >> 4;
    int nt = tile % nTiles;
    int kt = tile / nTiles;
    int n  = nt * 16 + (lane & 15);
    int k  = kt * 32 + ((lane < 16) ? 0 : 16) + i;
    float v = (k < Kd) ? W[(size_t)k * Nd + n] : 0.0f;
    out[idx] = (bf16)v;
}

// ---------------- LayerNorm (one wave per 256-wide row) ----------------
__global__ void ln_cast_kernel(const float* __restrict__ x, const float* __restrict__ w,
                               const float* __restrict__ b, bf16* __restrict__ out, int rows) {
    int wave = (int)((blockIdx.x * blockDim.x + threadIdx.x) >> 5);
    int lane = threadIdx.x & 31;
    if (wave >= rows) return;
    const float* xr = x + (size_t)wave * DMODEL;
    float v[8];
    float s = 0.0f;
#pragma unroll
    for (int i = 0; i < 8; ++i) { v[i] = xr[lane * 8 + i]; s += v[i]; }
#pragma unroll
    for (int off = 16; off; off >>= 1) s += __shfl_xor(s, off, 32);
    float mean = s * (1.0f / DMODEL);
    float q = 0.0f;
#pragma unroll
    for (int i = 0; i < 8; ++i) { float d = v[i] - mean; q += d * d; }
#pragma unroll
    for (int off = 16; off; off >>= 1) q += __shfl_xor(q, off, 32);
    float rstd = rsqrtf(q * (1.0f / DMODEL) + 1e-5f);
    bf16* orow = out + (size_t)wave * DMODEL;
#pragma unroll
    for (int i = 0; i < 8; ++i) {
        int c = lane * 8 + i;
        orow[c] = (bf16)((v[i] - mean) * rstd * w[c] + b[c]);
    }
}

__global__ void ln_f32_kernel(const float* __restrict__ x, const float* __restrict__ w,
                              const float* __restrict__ b, float* __restrict__ out, int rows) {
    int wave = (int)((blockIdx.x * blockDim.x + threadIdx.x) >> 5);
    int lane = threadIdx.x & 31;
    if (wave >= rows) return;
    const float* xr = x + (size_t)wave * DMODEL;
    float v[8];
    float s = 0.0f;
#pragma unroll
    for (int i = 0; i < 8; ++i) { v[i] = xr[lane * 8 + i]; s += v[i]; }
#pragma unroll
    for (int off = 16; off; off >>= 1) s += __shfl_xor(s, off, 32);
    float mean = s * (1.0f / DMODEL);
    float q = 0.0f;
#pragma unroll
    for (int i = 0; i < 8; ++i) { float d = v[i] - mean; q += d * d; }
#pragma unroll
    for (int off = 16; off; off >>= 1) q += __shfl_xor(q, off, 32);
    float rstd = rsqrtf(q * (1.0f / DMODEL) + 1e-5f);
    float* orow = out + (size_t)wave * DMODEL;
#pragma unroll
    for (int i = 0; i < 8; ++i) {
        int c = lane * 8 + i;
        orow[c] = (v[i] - mean) * rstd * w[c] + b[c];
    }
}

// ---------------- WMMA GEMM, NB N-tiles per wave ----------------------------
// C[M,N] = A[M,K](bf16,row-major) x Bpacked + epilogue.
// EPI: 0 plain store, 1 +bias then softplus, 2 +residual(aux), 3 +bias(aux)
// NB independent accumulators per wave -> independent WMMA chains (hides the
// WMMA->WMMA RAW hazard) and reuses the A fragment NB times.
template <int EPI, int NB>
__global__ void gemm_wmma(const bf16* __restrict__ A, const bf16* __restrict__ Bp,
                          float* __restrict__ C, const float* __restrict__ aux,
                          int Mt, int Nt, int ktiles, int lda, int ldc) {
    const int ngroups = Nt / NB;
    int wave = blockIdx.x * (blockDim.x >> 5) + (threadIdx.x >> 5);
    if (wave >= Mt * ngroups) return;            // uniform per-wave exit; EXEC all-1 for WMMA
    int mt  = wave / ngroups;
    int nt0 = (wave % ngroups) * NB;
    int lane = threadIdx.x & 31;
    int hl = lane & 15;
    int hi = lane >> 4;

    v8f acc[NB];
#pragma unroll
    for (int j = 0; j < NB; ++j)
#pragma unroll
        for (int i = 0; i < 8; ++i) acc[j][i] = 0.0f;

    const bf16* arow  = A + (size_t)(mt * 16 + hl) * lda + hi * 16;
    const bf16* bbase = Bp + ((size_t)nt0 * 32 + lane) * 16;
    const size_t bstride = (size_t)Nt * 512;     // Nt*32 lanes * 16 halves per k-tile

    for (int kt = 0; kt < ktiles; ++kt) {
        union { v16bf v; v8bf h[2]; } a;
        a.h[0] = *(const v8bf*)(arow);
        a.h[1] = *(const v8bf*)(arow + 8);
#pragma unroll
        for (int j = 0; j < NB; ++j) {
            union { v16bf v; v8bf h[2]; } bm;
            const bf16* bl = bbase + (size_t)j * 512;    // next n-tile: 32 lanes * 16 halves
            bm.h[0] = *(const v8bf*)(bl);
            bm.h[1] = *(const v8bf*)(bl + 8);
            acc[j] = __builtin_amdgcn_wmma_f32_16x16x32_bf16(false, a.v, false, bm.v,
                                                             (short)0, acc[j], false, false);
        }
        arow  += 32;
        bbase += bstride;
    }

    int rbase = mt * 16 + (hi ? 8 : 0);
#pragma unroll
    for (int j = 0; j < NB; ++j) {
        int col = (nt0 + j) * 16 + hl;
#pragma unroll
        for (int v = 0; v < 8; ++v) {
            int row = rbase + v;
            float val = acc[j][v];
            if (EPI == 1) {
                float t = val + aux[col];
                val = (t > 20.0f) ? t : log1pf(__expf(t));   // softplus
            } else if (EPI == 2) {
                val += aux[(size_t)row * ldc + col];         // residual add
            } else if (EPI == 3) {
                val += aux[col];                             // bias add
            }
            C[(size_t)row * ldc + col] = val;
        }
    }
}

// ---------------- causal depthwise conv (K=4) + bias + SiLU ----------------
__global__ void conv_silu_kernel(const float* __restrict__ inproj, const float* __restrict__ cw,
                                 const float* __restrict__ cb, float* __restrict__ uc,
                                 bf16* __restrict__ ucb) {
    int idx = blockIdx.x * blockDim.x + threadIdx.x;
    if (idx >= ROWS * DINNER) return;
    int d  = idx % DINNER;
    int bt = idx / DINNER;
    int t  = bt % TLEN;
    float s = cb[d];
#pragma unroll
    for (int k = 0; k < KCONV; ++k) {
        int tt = t - (KCONV - 1) + k;
        if (tt >= 0)
            s += inproj[(size_t)(bt - (KCONV - 1) + k) * (2 * DINNER) + d] * cw[d * KCONV + k];
    }
    float r = s * sigmoidf_(s);
    uc[idx]  = r;
    ucb[idx] = (bf16)r;
}

// ---------------- dt_low slice -> bf16, K padded 16 -> 32 ----------------
__global__ void dtlow_kernel(const float* __restrict__ proj, bf16* __restrict__ dtlow) {
    int idx = blockIdx.x * blockDim.x + threadIdx.x;
    if (idx >= ROWS * 32) return;
    int i = idx & 31;
    int r = idx >> 5;
    dtlow[idx] = (i < DTRANK) ? (bf16)proj[(size_t)r * 48 + i] : (bf16)0.0f;
}

// ---------------- selective scan: lane = state n, 2 channels per wave -------
// B_t / C_t (shared by all 8 waves of the block) are staged through LDS in
// 128-timestep chunks: one coalesced global pass replaces 8x redundant reads.
constexpr int SCHUNK = 128;
__global__ void scan_kernel(const float* __restrict__ dt, const float* __restrict__ uc,
                            const float* __restrict__ proj, const float* __restrict__ Alog,
                            const float* __restrict__ Dsk, float* __restrict__ y) {
    __shared__ float sBC[SCHUNK * 32];   // [t_local][0..15]=B, [16..31]=C
    int b    = blockIdx.x >> 5;          // 32 blocks per batch (DI/16)
    int cblk = blockIdx.x & 31;
    int w    = threadIdx.x >> 5;
    int lane = threadIdx.x & 31;
    int n    = lane & 15;
    int half = lane >> 4;
    int d    = cblk * 16 + w * 2 + half;

    float a     = -__expf(Alog[d * NST + n]);
    float dskip = Dsk[d];
    float h = 0.0f;
    size_t base = (size_t)b * TLEN;

    for (int tc = 0; tc < TLEN; tc += SCHUNK) {
        // cooperative stage of B/C for this chunk
        for (int i = threadIdx.x; i < SCHUNK * 32; i += 256) {
            int tl = i >> 5;
            int j  = i & 31;
            sBC[i] = proj[(base + tc + tl) * 48 + DTRANK + j];
        }
        __syncthreads();
        for (int tl = 0; tl < SCHUNK; ++tl) {
            size_t bt = base + tc + tl;
            float dtv = dt[bt * DINNER + d];
            float uv  = uc[bt * DINNER + d];
            float Bv  = sBC[tl * 32 + n];
            float Cv  = sBC[tl * 32 + 16 + n];
            h = __expf(dtv * a) * h + (dtv * uv) * Bv;
            float p = h * Cv;
            p += __shfl_xor(p, 8, 32);
            p += __shfl_xor(p, 4, 32);
            p += __shfl_xor(p, 2, 32);
            p += __shfl_xor(p, 1, 32);
            if (n == 0) y[bt * DINNER + d] = p + uv * dskip;
        }
        __syncthreads();
    }
}

// ---------------- gate: yg = y * silu(res) -> bf16 ----------------
__global__ void gate_kernel(const float* __restrict__ inproj, const float* __restrict__ y,
                            bf16* __restrict__ ygb) {
    int idx = blockIdx.x * blockDim.x + threadIdx.x;
    if (idx >= ROWS * DINNER) return;
    int d  = idx % DINNER;
    int bt = idx / DINNER;
    float res = inproj[(size_t)bt * (2 * DINNER) + DINNER + d];
    float g = res * sigmoidf_(res);
    ygb[idx] = (bf16)(y[idx] * g);
}

// ---------------- fp32 -> bf16 cast ----------------
__global__ void cast_kernel(const float* __restrict__ x, bf16* __restrict__ o, int n) {
    int idx = blockIdx.x * blockDim.x + threadIdx.x;
    if (idx < n) o[idx] = (bf16)x[idx];
}

// ---------------- mean over T (fixed order -> deterministic) ----------------
__global__ void mean_kernel(const float* __restrict__ tmp, float* __restrict__ out) {
    int idx = blockIdx.x * blockDim.x + threadIdx.x;
    if (idx >= BATCH * DMODEL) return;
    int d = idx % DMODEL;
    int b = idx / DMODEL;
    float s = 0.0f;
    for (int t = 0; t < TLEN; ++t)
        s += tmp[((size_t)b * TLEN + t) * DMODEL + d];
    out[idx] = s * (1.0f / TLEN);
}

// ---------------- host launcher ----------------
extern "C" void kernel_launch(void* const* d_in, const int* in_sizes, int n_in,
                              void* d_out, int out_size, void* d_ws, size_t ws_size,
                              hipStream_t stream) {
    (void)in_sizes; (void)n_in; (void)out_size; (void)ws_size;
    const float* x_in    = (const float*)d_in[0];
    const float* ln_w    = (const float*)d_in[1];
    const float* ln_b    = (const float*)d_in[2];
    const float* in_w    = (const float*)d_in[3];
    const float* conv_w  = (const float*)d_in[4];
    const float* conv_b  = (const float*)d_in[5];
    const float* xproj_w = (const float*)d_in[6];
    const float* dt_w    = (const float*)d_in[7];
    const float* dt_b    = (const float*)d_in[8];
    const float* A_log   = (const float*)d_in[9];
    const float* D_skip  = (const float*)d_in[10];
    const float* out_w   = (const float*)d_in[11];
    const float* oproj_w = (const float*)d_in[12];
    const float* oproj_b = (const float*)d_in[13];
    const float* fln_w   = (const float*)d_in[14];
    const float* fln_b   = (const float*)d_in[15];
    float* out = (float*)d_out;

    char* ws = (char*)d_ws;
    size_t off = 0;
    auto alloc = [&](size_t bytes) -> char* {
        char* p = ws + off;
        off = (off + bytes + 255) & ~(size_t)255;
        return p;
    };

    // packed weights (bf16)
    bf16* pk_in    = (bf16*)alloc((size_t)LNUM * 256 * 1024 * 2);
    bf16* pk_xproj = (bf16*)alloc((size_t)LNUM * 512 * 48 * 2);
    bf16* pk_dt    = (bf16*)alloc((size_t)LNUM * 32 * 512 * 2);
    bf16* pk_out   = (bf16*)alloc((size_t)LNUM * 512 * 256 * 2);
    bf16* pk_oproj = (bf16*)alloc((size_t)256 * 256 * 2);
    // activations
    bf16*  xn_bf   = (bf16*)alloc((size_t)ROWS * DMODEL * 2);
    float* inproj  = (float*)alloc((size_t)ROWS * 1024 * 4);
    float* uc      = (float*)alloc((size_t)ROWS * DINNER * 4);
    bf16*  ucb     = (bf16*)alloc((size_t)ROWS * DINNER * 2);
    float* proj    = (float*)alloc((size_t)ROWS * 48 * 4);
    bf16*  dtlowb  = (bf16*)alloc((size_t)ROWS * 32 * 2);
    float* dtbuf   = (float*)alloc((size_t)ROWS * DINNER * 4);
    float* ybuf    = (float*)alloc((size_t)ROWS * DINNER * 4);
    bf16*  ygb     = (bf16*)alloc((size_t)ROWS * DINNER * 2);
    float* x_ws    = (float*)alloc((size_t)ROWS * DMODEL * 4);
    float* xo      = (float*)alloc((size_t)ROWS * DMODEL * 4);
    float* lntmp   = (float*)alloc((size_t)ROWS * DMODEL * 4);

    auto g = [](int total) { return (total + 255) / 256; };

    // ---- pack all weight matrices ----
    for (int l = 0; l < LNUM; ++l) {
        pack_b_kernel<<<g(8 * 32 * 1024), 256, 0, stream>>>(
            in_w + (size_t)l * 256 * 1024, pk_in + (size_t)l * 256 * 1024, 256, 1024, 8);
        pack_b_kernel<<<g(16 * 32 * 48), 256, 0, stream>>>(
            xproj_w + (size_t)l * 512 * 48, pk_xproj + (size_t)l * 512 * 48, 512, 48, 16);
        pack_b_kernel<<<g(1 * 32 * 512), 256, 0, stream>>>(
            dt_w + (size_t)l * 16 * 512, pk_dt + (size_t)l * 32 * 512, 16, 512, 1);
        pack_b_kernel<<<g(16 * 32 * 256), 256, 0, stream>>>(
            out_w + (size_t)l * 512 * 256, pk_out + (size_t)l * 512 * 256, 512, 256, 16);
    }
    pack_b_kernel<<<g(8 * 32 * 256), 256, 0, stream>>>(oproj_w, pk_oproj, 256, 256, 8);

    // ---- layers ----
    for (int l = 0; l < LNUM; ++l) {
        const float* xsrc = (l == 0) ? x_in : x_ws;
        ln_cast_kernel<<<ROWS / 8, 256, 0, stream>>>(xsrc, ln_w + l * DMODEL, ln_b + l * DMODEL,
                                                     xn_bf, ROWS);
        // in-projection: 4096x256 -> 1024   (Nt=64, NB=4 -> 4096 waves)
        gemm_wmma<0, 4><<<(MT * 16) / 8, 256, 0, stream>>>(
            xn_bf, pk_in + (size_t)l * 256 * 1024, inproj, nullptr, MT, 64, 8, 256, 1024);
        conv_silu_kernel<<<g(ROWS * DINNER), 256, 0, stream>>>(
            inproj, conv_w + (size_t)l * DINNER * KCONV, conv_b + (size_t)l * DINNER, uc, ucb);
        // x-projection: 4096x512 -> 48      (Nt=3, NB=3 -> 256 waves)
        gemm_wmma<0, 3><<<(MT * 1) / 8, 256, 0, stream>>>(
            ucb, pk_xproj + (size_t)l * 512 * 48, proj, nullptr, MT, 3, 16, 512, 48);
        dtlow_kernel<<<g(ROWS * 32), 256, 0, stream>>>(proj, dtlowb);
        // dt-projection + softplus: 4096x32(pad) -> 512  (Nt=32, NB=4)
        gemm_wmma<1, 4><<<(MT * 8) / 8, 256, 0, stream>>>(
            dtlowb, pk_dt + (size_t)l * 32 * 512, dtbuf, dt_b + (size_t)l * DINNER,
            MT, 32, 1, 32, 512);
        scan_kernel<<<BATCH * (DINNER / 16), 256, 0, stream>>>(
            dtbuf, uc, proj, A_log + (size_t)l * DINNER * NST, D_skip + (size_t)l * DINNER, ybuf);
        gate_kernel<<<g(ROWS * DINNER), 256, 0, stream>>>(inproj, ybuf, ygb);
        // out-projection + residual: 4096x512 -> 256  (Nt=16, NB=4)
        gemm_wmma<2, 4><<<(MT * 4) / 8, 256, 0, stream>>>(
            ygb, pk_out + (size_t)l * 512 * 256, x_ws, xsrc, MT, 16, 16, 512, 256);
    }

    // ---- head: oproj + bias, final LN, mean over T ----
    cast_kernel<<<g(ROWS * DMODEL), 256, 0, stream>>>(x_ws, xn_bf, ROWS * DMODEL);
    gemm_wmma<3, 4><<<(MT * 4) / 8, 256, 0, stream>>>(
        xn_bf, pk_oproj, xo, oproj_b, MT, 16, 8, 256, 256);
    ln_f32_kernel<<<ROWS / 8, 256, 0, stream>>>(xo, fln_w, fln_b, lntmp, ROWS);
    mean_kernel<<<g(BATCH * DMODEL), 256, 0, stream>>>(lntmp, out);
}
